// CatNet_55027120996358
// MI455X (gfx1250) — matrix operand order repaired and checked by reading, not compiled, not run on hardware
//
#include <hip/hip_runtime.h>

typedef __attribute__((ext_vector_type(16))) _Float16 v16h;
typedef __attribute__((ext_vector_type(8)))  _Float16 v8h;
typedef __attribute__((ext_vector_type(8)))  float    v8f;

#define THETA 1.0f

// ---------------------------------------------------------------------------
// Fragment helpers (CDNA5 wave32 WMMA 16x16x32 f16, f32 accumulate)
//
// A (16x32 f16): lane l<16 holds row M=l, K=0..7 (VGPR0-3) and K=16..23
// (VGPR4-7); lane l+16 holds row M=l, K=8..15 and K=24..31.  With A rows
// stored as 32 contiguous halves, each lane loads from rowbase + (hi?8:0)
// and rowbase + (hi?8:0) + 16.
//
// B (32x16 f16): lane n<16 holds column n, K=0..7 and K=16..23; lane n+16 holds column
// n, K=8..15 and K=24..31 (same order as A); 2 K-values per VGPR.  We pre-pack B so each lane just loads
// 16 contiguous halves at bpack + lane*16.
// ---------------------------------------------------------------------------
__device__ __forceinline__ v16h load_a_frag(const _Float16* p /* rowbase + hi*8 */) {
  v8h lo = *(const v8h*)p;
  v8h hh = *(const v8h*)(p + 16);
  v16h a;
#pragma unroll
  for (int j = 0; j < 8; ++j) { a[j] = lo[j]; a[j + 8] = hh[j]; }
  return a;
}

__device__ __forceinline__ v8f wmma_f16(v16h a, v16h b, v8f c) {
  return __builtin_amdgcn_wmma_f32_16x16x32_f16(false, a, false, b, (short)0, c,
                                                false, false);
}

// ---------------------------------------------------------------------------
// Zero the 1-pixel border of a padded [2048][(W+2)][(W+2)][C] f16 buffer.
// ---------------------------------------------------------------------------
__global__ void zero_border(_Float16* __restrict__ buf, int W, int C) {
  const int PW = W + 2;
  const int nb = 2 * PW + 2 * W;               // border pixels per image
  const long total = 2048L * nb * C;
  long i = (long)blockIdx.x * blockDim.x + threadIdx.x;
  if (i >= total) return;
  int c = (int)(i % C);
  long r = i / C;
  int bp = (int)(r % nb);
  long img = r / nb;
  int pix;
  if (bp < PW)            pix = bp;                                 // top row
  else if (bp < 2 * PW)   pix = (W + 1) * PW + (bp - PW);           // bottom row
  else { int q = bp - 2 * PW; pix = (q / 2 + 1) * PW + ((q & 1) ? (W + 1) : 0); }
  buf[img * (long)(PW * PW * C) + (long)pix * C + c] = (_Float16)0.0f;
}

// ---------------------------------------------------------------------------
// Repack f32 weights into the exact per-lane B-fragment layout.
// dst index = ((chunk*npanels + panel)*32 + lane)*16 + j
// lane's K index = (lane>=16 ? 8 : 0) + (j&7) + (j>=8 ? 16 : 0) ; output channel o = panel*16 + (lane&15)
// modes: 0 conv 3x3 Cin=32 ([O][32][3][3])     1 conv 1x1 Cin=32 ([O][32])
//        2 conv 3x3 Cin=16 tap-paired          3 conv 1x1 Cin=16 (upper K zero)
//        4 dense1 ([128][6272], K=p*32+c)      5 dense2 ([10][128], o>=10 -> 0)
// ---------------------------------------------------------------------------
__global__ void pack_b(const float* __restrict__ w, _Float16* __restrict__ dst,
                       int nchunks, int npanels, int mode) {
  int i = blockIdx.x * 256 + threadIdx.x;
  int total = nchunks * npanels * 512;
  if (i >= total) return;
  int j = i & 15;
  int lane = (i >> 4) & 31;
  int rest = i >> 9;
  int panel = rest % npanels;
  int chunk = rest / npanels;
  int o = panel * 16 + (lane & 15);
  int k = ((lane >> 4) << 3) + (j & 7) + ((j >> 3) << 4);  // 0..31
  float v = 0.0f;
  switch (mode) {
    case 0: { int kh = chunk / 3, kw = chunk % 3;
              v = w[(o * 32 + k) * 9 + kh * 3 + kw]; } break;
    case 1: v = w[o * 32 + k]; break;
    case 2: { int kh = chunk >> 1, pr = chunk & 1;
              int kw = pr * 2 + (k >> 4); int c = k & 15;
              v = (kw < 3) ? w[(o * 16 + c) * 9 + kh * 3 + kw] : 0.0f; } break;
    case 3: v = (k < 16) ? w[o * 16 + k] : 0.0f; break;
    case 4: v = w[(long)o * 6272 + k * 196 + chunk]; break;
    case 5: v = (o < 10) ? w[o * 128 + chunk * 32 + k] : 0.0f; break;
  }
  dst[i] = (_Float16)v;
}

// ---------------------------------------------------------------------------
// Layer 1: scalar 3x3 conv (Cin=1) fused with the LIF scan.
// x: [128][1][28][28][16] f32 (t innermost).  Thread = (n, pixel, o-group of 8).
// Writes spikes into padded [img][30][30][32] f16.
// ---------------------------------------------------------------------------
__launch_bounds__(256)
__global__ void conv1_lif(const float* __restrict__ x, const float* __restrict__ w1,
                          const float* __restrict__ b1, _Float16* __restrict__ s1) {
  int idx = blockIdx.x * 256 + threadIdx.x;
  if (idx >= 128 * 784 * 4) return;
  int g = idx & 3;
  int rest = idx >> 2;
  int p = rest % 784, n = rest / 784;
  int h = p / 28, w = p % 28;

  long off[9]; bool val[9];
#pragma unroll
  for (int tap = 0; tap < 9; ++tap) {
    int hh = h + tap / 3 - 1, ww = w + tap % 3 - 1;
    bool v = (hh >= 0) & (hh < 28) & (ww >= 0) & (ww < 28);
    val[tap] = v;
    off[tap] = v ? ((long)(n * 784 + hh * 28 + ww) * 16) : 0;
  }
  float wt[8][9], u[8], bb[8];
#pragma unroll
  for (int o8 = 0; o8 < 8; ++o8) {
    bb[o8] = b1[g * 8 + o8];
    u[o8] = 0.0f;
#pragma unroll
    for (int tap = 0; tap < 9; ++tap) wt[o8][tap] = w1[(g * 8 + o8) * 9 + tap];
  }
  const long pixoff = (long)((h + 1) * 30 + (w + 1)) * 32 + g * 8;
  for (int t = 0; t < 16; ++t) {
    float xv[9];
#pragma unroll
    for (int tap = 0; tap < 9; ++tap) xv[tap] = val[tap] ? x[off[tap] + t] : 0.0f;
    _Float16* dst = s1 + (long)(n * 16 + t) * 28800 + pixoff;
#pragma unroll
    for (int o8 = 0; o8 < 8; ++o8) {
      float a = bb[o8];
#pragma unroll
      for (int tap = 0; tap < 9; ++tap) a += wt[o8][tap] * xv[tap];
      u[o8] += a;
      float s = (u[o8] >= THETA) ? 1.0f : 0.0f;
      u[o8] -= s;
      dst[o8] = (_Float16)s;
    }
  }
}

// ---------------------------------------------------------------------------
// Generic fused WMMA conv + LIF, fully templated so the K-chunk loop unrolls
// with compile-time tap offsets (no per-iteration SALU control flow).
// One wave per block; block owns (n, 32-pixel tile, all Cout, all T).
// Per t: implicit-GEMM via WMMA (2 M-frags x npanels) -> potentials into LDS;
// then the 16-step LIF scan runs from LDS and emits f16 spikes.
// MODE: 0 = 3x3 Cin32 (9 chunks)   1 = 1x1 Cin32 (1 chunk)
//       2 = 3x3 Cin16 paired taps  3 = 1x1 Cin16 (A reads 2 pixels, upper K w=0)
// ---------------------------------------------------------------------------
struct ConvP {
  int W, PWin, npix;
  long in_stride, out_stride, res_stride;   // halves per image
  int out_PW, out_pad, res_PW;
};

template <int MODE> __device__ __forceinline__ constexpr int nchunks_of() {
  return MODE == 0 ? 9 : (MODE == 2 ? 6 : 1);
}
template <int MODE> __device__ __forceinline__ void chunk_dhdw(int ck, int& dh, int& dw) {
  if (MODE == 0)      { dh = ck / 3;  dw = ck % 3; }
  else if (MODE == 2) { dh = ck >> 1; dw = (ck & 1) * 2; }
  else                { dh = 1;       dw = 1; }
}

template <int MODE, int CIN, int COUT, bool HAS_RES>
__launch_bounds__(32)
__global__ void conv_lif(const _Float16* __restrict__ sin,
                         const _Float16* __restrict__ bpack,
                         const float* __restrict__ bias,
                         const _Float16* __restrict__ res,
                         _Float16* __restrict__ sout, ConvP P) {
  constexpr int NCK = nchunks_of<MODE>();
  constexpr int NPAN = COUT / 16;
  __shared__ float pot[16][32 * COUT];

  const int lane = threadIdx.x;
  const int hi = lane >> 4, m = lane & 15;
  const int tile = blockIdx.x, n = blockIdx.y;

  // two 16-row M-fragments per block (pixels tile*32+m and tile*32+16+m)
  int p0 = tile * 32 + m;       if (p0 >= P.npix) p0 = P.npix - 1;
  int p1 = tile * 32 + 16 + m;  if (p1 >= P.npix) p1 = P.npix - 1;
  const int h0 = p0 / P.W, w0 = p0 % P.W;
  const int h1 = p1 / P.W, w1 = p1 % P.W;

  // per-lane A offsets for every chunk (halves), hoisted out of the t loop
  size_t aoff0[NCK], aoff1[NCK];
#pragma unroll
  for (int ck = 0; ck < NCK; ++ck) {
    int dh, dw;
    chunk_dhdw<MODE>(ck, dh, dw);
    aoff0[ck] = (size_t)((h0 + dh) * P.PWin + (w0 + dw)) * CIN + hi * 8;
    aoff1[ck] = (size_t)((h1 + dh) * P.PWin + (w1 + dw)) * CIN + hi * 8;
  }
  const float bv0 = bias[m];
  const float bv1 = (NPAN == 2) ? bias[16 + m] : 0.0f;

  for (int t = 0; t < 16; ++t) {
    const _Float16* abase = sin + (size_t)(n * 16 + t) * P.in_stride;
    v8f acc00 = {bv0, bv0, bv0, bv0, bv0, bv0, bv0, bv0};  // tile0, panel0
    v8f acc10 = acc00;                                     // tile1, panel0
    v8f acc01 = {bv1, bv1, bv1, bv1, bv1, bv1, bv1, bv1};  // tile0, panel1
    v8f acc11 = acc01;                                     // tile1, panel1
#pragma unroll
    for (int ck = 0; ck < NCK; ++ck) {
      v16h a0 = load_a_frag(abase + aoff0[ck]);
      v16h a1 = load_a_frag(abase + aoff1[ck]);
      const _Float16* bp = bpack + (size_t)(ck * NPAN) * 512 + lane * 16;
      v16h b0 = *(const v16h*)bp;
      acc00 = wmma_f16(a0, b0, acc00);
      acc10 = wmma_f16(a1, b0, acc10);
      if (NPAN == 2) {
        v16h b1 = *(const v16h*)(bp + 512);
        acc01 = wmma_f16(a0, b1, acc01);
        acc11 = wmma_f16(a1, b1, acc11);
      }
    }
    // D layout: lane l, elem r -> M = r + (l>=16 ? 8 : 0), N = l&15
#pragma unroll
    for (int r = 0; r < 8; ++r) {
      const int pm0 = r + hi * 8, pm1 = 16 + pm0;
      pot[t][pm0 * COUT + m] = acc00[r];
      pot[t][pm1 * COUT + m] = acc10[r];
      if (NPAN == 2) {
        pot[t][pm0 * COUT + 16 + m] = acc01[r];
        pot[t][pm1 * COUT + 16 + m] = acc11[r];
      }
    }
  }
  __syncthreads();

  // LIF scan: each lane owns neurons (pixel-in-tile, channel)
  constexpr int TOT = 32 * COUT;
#pragma unroll 1
  for (int idx = lane; idx < TOT; idx += 32) {
    const int pm = idx / COUT, ch = idx % COUT;   // COUT is a power of two
    const int pp = tile * 32 + pm;
    const bool valid = pp < P.npix;
    const int oh = pp / P.W, ow = pp % P.W;
    size_t obase = P.out_pad
        ? (size_t)((oh + 1) * P.out_PW + (ow + 1)) * COUT + ch
        : (size_t)pp * COUT + ch;
    size_t rbase = 0;
    if (HAS_RES) rbase = (size_t)((oh + 1) * P.res_PW + (ow + 1)) * COUT + ch;
    float u = 0.0f;
#pragma unroll
    for (int t = 0; t < 16; ++t) {
      const size_t img = (size_t)(n * 16 + t);
      float xv = pot[t][pm * COUT + ch];
      if (HAS_RES) xv += (float)res[img * P.res_stride + rbase];
      u += xv;
      float s = (u >= THETA) ? 1.0f : 0.0f;
      u -= s;
      if (valid) sout[img * P.out_stride + obase] = (_Float16)s;
    }
  }
}

// ---------------------------------------------------------------------------
// 2x2 average pool + LIF.  s3 padded [img][30][30][32] -> sp padded [img][16][16][32].
// ---------------------------------------------------------------------------
__launch_bounds__(256)
__global__ void pool_lif(const _Float16* __restrict__ s3, _Float16* __restrict__ sp) {
  int idx = blockIdx.x * 256 + threadIdx.x;
  if (idx >= 128 * 196 * 32) return;
  int c = idx & 31;
  int r = idx >> 5;
  int p = r % 196, n = r / 196;
  int h = p / 14, w = p % 14;
  const size_t i00 = (size_t)((2 * h + 1) * 30 + (2 * w + 1)) * 32 + c;
  const size_t dsto = (size_t)((h + 1) * 16 + (w + 1)) * 32 + c;
  float u = 0.0f;
#pragma unroll
  for (int t = 0; t < 16; ++t) {
    const size_t img = (size_t)(n * 16 + t);
    const _Float16* b = s3 + img * 28800;
    float v = (float)b[i00] + (float)b[i00 + 32] +
              (float)b[i00 + 30 * 32] + (float)b[i00 + 31 * 32];
    u += 0.25f * v;
    float s = (u >= THETA) ? 1.0f : 0.0f;
    u -= s;
    sp[img * 8192 + dsto] = (_Float16)s;
  }
}

// ---------------------------------------------------------------------------
// Dense1 (K=6272, O=128) + LIF.  Block = one n; its 16 M-rows are exactly the
// 16 timesteps of that n, so the scan stays in-block.  4 waves x 2 panels.
// ---------------------------------------------------------------------------
__launch_bounds__(128)
__global__ void dense1_lif(const _Float16* __restrict__ s6,
                           const _Float16* __restrict__ bpack,
                           const float* __restrict__ bf1,
                           _Float16* __restrict__ sf1) {
  __shared__ float pot[16][128];
  const int n = blockIdx.x;
  const int lane = threadIdx.x & 31, wv = threadIdx.x >> 5;
  const int hi = lane >> 4, m = lane & 15;
  const _Float16* arow = s6 + (size_t)(n * 16 + m) * 6272 + hi * 8;
  const float bv0 = bf1[wv * 32 + m], bv1 = bf1[wv * 32 + 16 + m];
  v8f acc0 = {bv0, bv0, bv0, bv0, bv0, bv0, bv0, bv0};
  v8f acc1 = {bv1, bv1, bv1, bv1, bv1, bv1, bv1, bv1};
#pragma unroll 4
  for (int q = 0; q < 196; ++q) {
    v16h a = load_a_frag(arow + q * 32);
    const _Float16* bp = bpack + (size_t)(q * 8 + wv * 2) * 512 + lane * 16;
    v16h b0 = *(const v16h*)bp;
    v16h b1 = *(const v16h*)(bp + 512);
    acc0 = wmma_f16(a, b0, acc0);
    acc1 = wmma_f16(a, b1, acc1);
  }
#pragma unroll
  for (int r = 0; r < 8; ++r) {
    pot[r + hi * 8][wv * 32 + m] = acc0[r];
    pot[r + hi * 8][wv * 32 + 16 + m] = acc1[r];
  }
  __syncthreads();
  const int o = threadIdx.x;
  float u = 0.0f;
#pragma unroll
  for (int t = 0; t < 16; ++t) {
    u += pot[t][o];
    float s = (u >= THETA) ? 1.0f : 0.0f;
    u -= s;
    sf1[(size_t)(n * 16 + t) * 128 + o] = (_Float16)s;
  }
}

// ---------------------------------------------------------------------------
// Dense2 (K=128, O=10 padded to 16) + LIF + firing-rate output [128][10] f32.
// ---------------------------------------------------------------------------
__launch_bounds__(32)
__global__ void dense2_out(const _Float16* __restrict__ sf1,
                           const _Float16* __restrict__ bpack,
                           const float* __restrict__ bf2,
                           float* __restrict__ out) {
  __shared__ float pot[16][16];
  const int n = blockIdx.x;
  const int lane = threadIdx.x;
  const int hi = lane >> 4, m = lane & 15;
  const _Float16* arow = sf1 + (size_t)(n * 16 + m) * 128 + hi * 8;
  const float bv = (m < 10) ? bf2[m] : 0.0f;
  v8f acc = {bv, bv, bv, bv, bv, bv, bv, bv};
#pragma unroll
  for (int q = 0; q < 4; ++q) {
    v16h a = load_a_frag(arow + q * 32);
    v16h b = *(const v16h*)(bpack + (size_t)q * 512 + lane * 16);
    acc = wmma_f16(a, b, acc);
  }
#pragma unroll
  for (int r = 0; r < 8; ++r) pot[r + hi * 8][m] = acc[r];
  __syncthreads();
  if (lane < 10) {
    float u = 0.0f, sum = 0.0f;
#pragma unroll
    for (int t = 0; t < 16; ++t) {
      u += pot[t][lane];
      float s = (u >= THETA) ? 1.0f : 0.0f;
      u -= s;
      sum += s;
    }
    out[n * 10 + lane] = sum * (1.0f / 16.0f);
  }
}

// ---------------------------------------------------------------------------
// Host launch
// ---------------------------------------------------------------------------
extern "C" void kernel_launch(void* const* d_in, const int* in_sizes, int n_in,
                              void* d_out, int out_size, void* d_ws, size_t ws_size,
                              hipStream_t stream) {
  (void)in_sizes; (void)n_in; (void)out_size; (void)ws_size;
  const float* x   = (const float*)d_in[0];
  const float* w1  = (const float*)d_in[1];  const float* b1  = (const float*)d_in[2];
  const float* w2  = (const float*)d_in[3];  const float* b2  = (const float*)d_in[4];
  const float* w3  = (const float*)d_in[5];  const float* b3  = (const float*)d_in[6];
  const float* w4  = (const float*)d_in[7];  const float* b4  = (const float*)d_in[8];
  const float* w5  = (const float*)d_in[9];  const float* b5  = (const float*)d_in[10];
  const float* w6  = (const float*)d_in[11]; const float* b6  = (const float*)d_in[12];
  const float* wf1 = (const float*)d_in[13]; const float* bf1 = (const float*)d_in[14];
  const float* wf2 = (const float*)d_in[15]; const float* bf2 = (const float*)d_in[16];
  float* out = (float*)d_out;

  char* ws = (char*)d_ws;
  const size_t SZ_S28  = 2048ULL * 28800 * 2;  // padded 30x30x32 f16
  const size_t SZ_SP   = 2048ULL * 8192 * 2;   // padded 16x16x32 f16
  const size_t SZ_S14  = 2048ULL * 4096 * 2;   // padded 16x16x16 f16
  const size_t SZ_S6   = 2048ULL * 6272 * 2;   // [196][32] f16 (unpadded)
  const size_t SZ_SF1  = 2048ULL * 128 * 2;
  size_t off = 0;
  _Float16* S1  = (_Float16*)(ws + off); off += SZ_S28;
  _Float16* S2  = (_Float16*)(ws + off); off += SZ_S28;
  _Float16* S3  = (_Float16*)(ws + off); off += SZ_S28;
  _Float16* SP  = (_Float16*)(ws + off); off += SZ_SP;
  _Float16* S4  = (_Float16*)(ws + off); off += SZ_S14;
  _Float16* S5  = (_Float16*)(ws + off); off += SZ_S14;
  _Float16* S6  = (_Float16*)(ws + off); off += SZ_S6;
  _Float16* SF1 = (_Float16*)(ws + off); off += SZ_SF1;
  _Float16* PK2 = (_Float16*)(ws + off); off += 9216 * 2;
  _Float16* PK3 = (_Float16*)(ws + off); off += 9216 * 2;
  _Float16* PK4 = (_Float16*)(ws + off); off += 512 * 2;
  _Float16* PK5 = (_Float16*)(ws + off); off += 3072 * 2;
  _Float16* PK6 = (_Float16*)(ws + off); off += 1024 * 2;
  _Float16* PD1 = (_Float16*)(ws + off); off += 802816ULL * 2;
  _Float16* PD2 = (_Float16*)(ws + off); off += 2048 * 2;

  // 1) zero padded borders (interiors are fully rewritten every call)
  {
    long t28 = 2048L * (2 * 30 + 2 * 28) * 32;
    long tsp = 2048L * (2 * 16 + 2 * 14) * 32;
    long t14 = 2048L * (2 * 16 + 2 * 14) * 16;
    zero_border<<<(t28 + 255) / 256, 256, 0, stream>>>(S1, 28, 32);
    zero_border<<<(t28 + 255) / 256, 256, 0, stream>>>(S2, 28, 32);
    zero_border<<<(t28 + 255) / 256, 256, 0, stream>>>(S3, 28, 32);
    zero_border<<<(tsp + 255) / 256, 256, 0, stream>>>(SP, 14, 32);
    zero_border<<<(t14 + 255) / 256, 256, 0, stream>>>(S4, 14, 16);
    zero_border<<<(t14 + 255) / 256, 256, 0, stream>>>(S5, 14, 16);
  }

  // 2) repack weights to WMMA B-fragment layout (f16)
  {
    auto pk = [&](const float* w, _Float16* dst, int nc, int np, int mode) {
      int tot = nc * np * 512;
      pack_b<<<(tot + 255) / 256, 256, 0, stream>>>(w, dst, nc, np, mode);
    };
    pk(w2, PK2, 9, 2, 0);
    pk(w3, PK3, 9, 2, 0);
    pk(w4, PK4, 1, 1, 1);
    pk(w5, PK5, 6, 1, 2);
    pk(w6, PK6, 1, 2, 3);
    pk(wf1, PD1, 196, 8, 4);
    pk(wf2, PD2, 4, 1, 5);
  }

  // 3) layer 1 (scalar conv + LIF)
  conv1_lif<<<(128 * 784 * 4 + 255) / 256, 256, 0, stream>>>(x, w1, b1, S1);

  // 4) WMMA conv layers (fused LIF), 32-pixel M-tiles
  ConvP L2{28, 30, 784, 28800, 28800, 0, 30, 1, 0};
  conv_lif<0, 32, 32, false><<<dim3(25, 128), 32, 0, stream>>>(S1, PK2, b2, nullptr, S2, L2);

  ConvP L3{28, 30, 784, 28800, 28800, 28800, 30, 1, 30};
  conv_lif<0, 32, 32, true><<<dim3(25, 128), 32, 0, stream>>>(S2, PK3, b3, S1, S3, L3);

  pool_lif<<<(128 * 196 * 32 + 255) / 256, 256, 0, stream>>>(S3, SP);

  ConvP L4{14, 16, 196, 8192, 4096, 0, 16, 1, 0};
  conv_lif<1, 32, 16, false><<<dim3(7, 128), 32, 0, stream>>>(SP, PK4, b4, nullptr, S4, L4);

  ConvP L5{14, 16, 196, 4096, 4096, 0, 16, 1, 0};
  conv_lif<2, 16, 16, false><<<dim3(7, 128), 32, 0, stream>>>(S4, PK5, b5, nullptr, S5, L5);

  ConvP L6{14, 16, 196, 4096, 6272, 8192, 14, 0, 16};
  conv_lif<3, 16, 32, true><<<dim3(7, 128), 32, 0, stream>>>(S5, PK6, b6, SP, S6, L6);

  // 5) dense layers (fused LIF) + firing-rate output
  dense1_lif<<<128, 128, 0, stream>>>(S6, PD1, bf1, SF1);
  dense2_out<<<128, 32, 0, stream>>>(SF1, PD2, bf2, out);
}
